// ProtoCADEpochShellModel_42253888258927
// MI455X (gfx1250) — matrix operand order repaired and checked by reading, not compiled, not run on hardware
//
#include <hip/hip_runtime.h>
#include <math.h>

typedef _Float16 v4h  __attribute__((ext_vector_type(4)));
typedef _Float16 v8h  __attribute__((ext_vector_type(8)));
typedef _Float16 v16h __attribute__((ext_vector_type(16)));
typedef float    v8f  __attribute__((ext_vector_type(8)));

#define Dk   1024
#define Bn   8192
#define TEMP 0.07f
#define ROWP 1032   // padded LDS row stride (f16 elems): 2064B = 516 dwords == 4 mod 64 banks

#define WMMA_F16(A, B, C) \
    __builtin_amdgcn_wmma_f32_16x16x32_f16(false, (A), false, (B), (short)0, (C), false, false)

// ---------------------------------------------------------------------------
// Kernel 1: per-row prep. One wave per row.
//   distances[i] = ||e_i - c||, Sp[i] = (e_i . c)/TEMP,
//   Ehi/Elo = split-f16 decomposition of E (x = hi + lo).
// ---------------------------------------------------------------------------
__global__ __launch_bounds__(256) void prep_kernel(
    const float* __restrict__ emb, const float* __restrict__ center,
    _Float16* __restrict__ Ehi, _Float16* __restrict__ Elo,
    float* __restrict__ Sp, float* __restrict__ dist)
{
    int wave = threadIdx.x >> 5, lane = threadIdx.x & 31;
    int row = blockIdx.x * 8 + wave;
    const float* e = emb + (size_t)row * Dk;

    float ss = 0.f, dot = 0.f;
#pragma unroll
    for (int t = 0; t < 8; ++t) {
        int idx = t * 128 + lane * 4;
        float4 v = *(const float4*)(e + idx);
        float4 c = *(const float4*)(center + idx);
        float d0 = v.x - c.x, d1 = v.y - c.y, d2 = v.z - c.z, d3 = v.w - c.w;
        ss  += d0 * d0 + d1 * d1 + d2 * d2 + d3 * d3;
        dot += v.x * c.x + v.y * c.y + v.z * c.z + v.w * c.w;
        _Float16 h0 = (_Float16)v.x, h1 = (_Float16)v.y,
                 h2 = (_Float16)v.z, h3 = (_Float16)v.w;
        v4h hv = { h0, h1, h2, h3 };
        v4h lv = { (_Float16)(v.x - (float)h0), (_Float16)(v.y - (float)h1),
                   (_Float16)(v.z - (float)h2), (_Float16)(v.w - (float)h3) };
        *(v4h*)(Ehi + (size_t)row * Dk + idx) = hv;
        *(v4h*)(Elo + (size_t)row * Dk + idx) = lv;
    }
#pragma unroll
    for (int m = 16; m; m >>= 1) {
        ss  += __shfl_xor(ss, m, 32);
        dot += __shfl_xor(dot, m, 32);
    }
    if (lane == 0) {
        dist[row] = sqrtf(ss);
        Sp[row]   = dot * (1.0f / TEMP);
    }
}

// ---------------------------------------------------------------------------
// Kernel 2: row-strip GEMM with split-f16 WMMA + fused row reductions.
// Block = 4 waves, owns 64 rows staged in LDS (hi+lo = 264KB -> 1 block/WGP,
// so VGPRs are free: each wave computes a 16x32 output slab per pass
// (2 j-tiles, 4 independent accumulators) vs all 8192 columns.
//   sumexp[i] = sum_{j!=i} exp(S[i,j]),  possum[i] = sum_{j!=i, lab_j==0} S[i,j]
// ---------------------------------------------------------------------------
__global__ __launch_bounds__(128) void gemm_kernel(
    const _Float16* __restrict__ Ehi, const _Float16* __restrict__ Elo,
    const int* __restrict__ labels,
    float* __restrict__ sumexp, float* __restrict__ possum)
{
    extern __shared__ _Float16 lds[];          // [2][64][ROWP]
    _Float16* Ah = lds;
    _Float16* Al = lds + 64 * ROWP;

    const int tid = threadIdx.x;
    const int i0  = blockIdx.x * 64;

    // cooperative stage of the 64-row A strip (hi and lo) into padded LDS
    for (int c = tid; c < 64 * 128; c += 128) {     // uint4 = 8 f16
        int r = c >> 7, col = c & 127;
        uint4 hv = *(const uint4*)(Ehi + (size_t)(i0 + r) * Dk + col * 8);
        uint4 lv = *(const uint4*)(Elo + (size_t)(i0 + r) * Dk + col * 8);
        *(uint4*)(Ah + r * ROWP + col * 8) = hv;
        *(uint4*)(Al + r * ROWP + col * 8) = lv;
    }
    __syncthreads();

    const int wave = tid >> 5, lane = tid & 31;
    const int sub = lane & 15, hg = lane >> 4;   // hg = half-group (lanes 0-15 / 16-31)
    const _Float16* ah_row = Ah + (wave * 16 + sub) * ROWP;
    const _Float16* al_row = Al + (wave * 16 + sub) * ROWP;

    float se[8] = {0.f}, ps[8] = {0.f};
    const float invT = 1.0f / TEMP;
    const int ibase = i0 + wave * 16 + hg * 8;

    for (int j0 = 0; j0 < Bn; j0 += 32) {
        // B-fragment: column n of E^T == row n of E; contiguous 16 K's per lane
        const _Float16* bh0 = Ehi + (size_t)(j0 + sub) * Dk;
        const _Float16* bl0 = Elo + (size_t)(j0 + sub) * Dk;
        const _Float16* bh1 = Ehi + (size_t)(j0 + 16 + sub) * Dk;
        const _Float16* bl1 = Elo + (size_t)(j0 + 16 + sub) * Dk;
        v8f c0a = {}, c1a = {}, c0b = {}, c1b = {};   // tile a: cols j0.., tile b: cols j0+16..
#pragma unroll 2
        for (int k0 = 0; k0 < Dk; k0 += 32) {
            v8h ah0 = *(const v8h*)(ah_row + k0 + hg * 16);       // K = k0+16hg .. +8
            v8h ah1 = *(const v8h*)(ah_row + k0 + hg * 16 + 8);   // K = k0+16hg+8 .. +8
            v8h al0 = *(const v8h*)(al_row + k0 + hg * 16);
            v8h al1 = *(const v8h*)(al_row + k0 + hg * 16 + 8);
            v16h bha = *(const v16h*)(bh0 + k0 + hg * 16);
            v16h bla = *(const v16h*)(bl0 + k0 + hg * 16);
            v16h bhb = *(const v16h*)(bh1 + k0 + hg * 16);
            v16h blb = *(const v16h*)(bl1 + k0 + hg * 16);
            v16h ah = __builtin_shufflevector(ah0, ah1, 0,1,2,3,4,5,6,7,8,9,10,11,12,13,14,15);
            v16h al = __builtin_shufflevector(al0, al1, 0,1,2,3,4,5,6,7,8,9,10,11,12,13,14,15);
            // interleave accumulators: consecutive WMMAs are independent
            c0a = WMMA_F16(ah, bha, c0a);
            c0b = WMMA_F16(ah, bhb, c0b);
            c1a = WMMA_F16(ah, bla, c1a);
            c1b = WMMA_F16(ah, blb, c1b);
            c1a = WMMA_F16(al, bha, c1a);
            c1b = WMMA_F16(al, bhb, c1b);
        }
        const int ja = j0 + sub, jb = j0 + 16 + sub;
        const int laba = labels[ja], labb = labels[jb];
#pragma unroll
        for (int r = 0; r < 8; ++r) {           // C layout: row = ibase + r, col = j
            float sa = (c0a[r] + c1a[r]) * invT;
            float sb = (c0b[r] + c1b[r]) * invT;
            bool da = (ibase + r) == ja;
            bool db = (ibase + r) == jb;
            se[r] += (da ? 0.f : __expf(sa)) + (db ? 0.f : __expf(sb));
            ps[r] += ((!da && laba == 0) ? sa : 0.f) + ((!db && labb == 0) ? sb : 0.f);
        }
    }
    // reduce the 16 column-lanes of each row; unique writer per row -> no atomics
#pragma unroll
    for (int r = 0; r < 8; ++r) {
        float a = se[r], b = ps[r];
#pragma unroll
        for (int m = 1; m < 16; m <<= 1) {
            a += __shfl_xor(a, m, 32);
            b += __shfl_xor(b, m, 32);
        }
        if (sub == 0) {
            int i = ibase + r;
            sumexp[i] = a;
            possum[i] = b;
        }
    }
}

// ---------------------------------------------------------------------------
// Kernel 3: scalar epilogue (single block).
// ---------------------------------------------------------------------------
__global__ __launch_bounds__(1024) void finalize_kernel(
    const float* __restrict__ Sp, const float* __restrict__ sumexp,
    const float* __restrict__ possum, const int* __restrict__ labels,
    const float* __restrict__ radius_m, const float* __restrict__ radius_h,
    const float* __restrict__ dist, float* __restrict__ out)
{
    __shared__ float sred[32][6];
    __shared__ float sfin[6];
    const int tid = threadIdx.x, lane = tid & 31, wave = tid >> 5;
    const float rm = radius_m[0], rh = radius_h[0];

    float acc[6] = {0.f, 0.f, 0.f, 0.f, 0.f, 0.f};  // nm, nh, lm, lh, sumexp_p, sum_Sp_m
    for (int i = tid; i < Bn; i += 1024) {
        int lab = labels[i];
        float d = dist[i], sp = Sp[i];
        if (lab == 0) {
            acc[0] += 1.f;
            float t = fmaxf(d - rm, 0.f); acc[2] += t * t;
            acc[5] += sp;
        } else {
            acc[1] += 1.f;
            float t = fmaxf(rh - d, 0.f); acc[3] += t * t;
        }
        acc[4] += __expf(sp);
    }
#pragma unroll
    for (int q = 0; q < 6; ++q) {
        float v = acc[q];
#pragma unroll
        for (int m = 16; m; m >>= 1) v += __shfl_xor(v, m, 32);
        if (lane == 0) sred[wave][q] = v;
    }
    __syncthreads();
    if (tid < 6) {
        float v = 0.f;
        for (int w = 0; w < 32; ++w) v += sred[w][tid];
        sfin[tid] = v;
    }
    __syncthreads();
    const float Nm = sfin[0], Nh = sfin[1], Lm = sfin[2], Lh = sfin[3],
                Sep = sfin[4], Ssm = sfin[5];
    const float nmd = fmaxf(Nm, 1.f);
    __syncthreads();

    float con = 0.f;
    for (int i = tid; i < Bn; i += 1024) {
        if (labels[i] == 0) {
            float lse = logf(sumexp[i] + __expf(Sp[i]));
            con += lse - (possum[i] + Sp[i]) / nmd;
        }
    }
#pragma unroll
    for (int m = 16; m; m >>= 1) con += __shfl_xor(con, m, 32);
    if (lane == 0) sred[wave][0] = con;
    __syncthreads();

    if (tid == 0) {
        float C = 0.f;
        for (int w = 0; w < 32; ++w) C += sred[w][0];
        float loss_m = Lm / fmaxf(Nm, 1.f);
        float loss_h = Lh / fmaxf(Nh, 1.f);
        float loss_shell = loss_m + loss_h;
        float proto = logf(Sep) - Ssm / nmd;
        float loss_con = (C + proto) / fmaxf(Nm + 1.f, 1.f);
        if (!((Nm > 0.f) && (Nh > 0.f))) loss_con = 0.f;
        float loss = loss_shell + loss_con;   // LAMBDA_SHELL = LAMBDA_CON = 1
        out[0] = loss; out[1] = loss_shell; out[2] = loss_m;
        out[3] = loss_h; out[4] = loss_con;
    }
}

// ---------------------------------------------------------------------------
extern "C" void kernel_launch(void* const* d_in, const int* in_sizes, int n_in,
                              void* d_out, int out_size, void* d_ws, size_t ws_size,
                              hipStream_t stream) {
    (void)in_sizes; (void)n_in; (void)out_size; (void)ws_size;
    const float* emb      = (const float*)d_in[0];
    const float* center   = (const float*)d_in[1];
    const float* radius_m = (const float*)d_in[2];
    const float* radius_h = (const float*)d_in[3];
    const int*   labels   = (const int*)d_in[4];
    float* out = (float*)d_out;

    char* ws = (char*)d_ws;
    _Float16* Ehi = (_Float16*)ws;                                  // 16 MB
    _Float16* Elo = (_Float16*)(ws + (size_t)Bn * Dk * 2);          // 16 MB
    float* Sp     = (float*)(ws + (size_t)Bn * Dk * 4);
    float* sumexp = Sp + Bn;
    float* possum = sumexp + Bn;

    // distances go straight into out[5..5+8191]
    prep_kernel<<<Bn / 8, 256, 0, stream>>>(emb, center, Ehi, Elo, Sp, out + 5);

    const int ldsBytes = 2 * 64 * ROWP * (int)sizeof(_Float16);     // 264192 B < 320 KB
    (void)hipFuncSetAttribute(reinterpret_cast<const void*>(gemm_kernel),
                              hipFuncAttributeMaxDynamicSharedMemorySize, ldsBytes);
    gemm_kernel<<<Bn / 64, 128, ldsBytes, stream>>>(Ehi, Elo, labels, sumexp, possum);

    finalize_kernel<<<1, 1024, 0, stream>>>(Sp, sumexp, possum, labels,
                                            radius_m, radius_h, out + 5, out);
}
